// ScaledDotProductAttention_32263794328193
// MI455X (gfx1250) — matrix operand order repaired and checked by reading, not compiled, not run on hardware
//
#include <hip/hip_runtime.h>
#include <hip/hip_bf16.h>
#include <math.h>

// Flash attention (causal), B=16, S=4096, DK=64, fp32 in/out.
// Transposed-score formulation (S^T = K x Q^T): softmax reductions are
// in-lane + one cross-half shuffle, and P^T's C-layout == P's A-layout.
// Row-sums of P via WMMA against an all-ones B fragment. Raw v_exp_f32 /
// v_rcp_f32. Ballot-guarded accumulator rescale. Double-buffered LDS with
// register-staged global loads: one barrier per key block, global latency
// hidden under the 18 WMMAs of the previous tile.

typedef __attribute__((ext_vector_type(16))) __bf16 v16bf;
typedef __attribute__((ext_vector_type(8)))  float  v8f;

#define B_    16
#define S_    4096
#define DK_   64
#define QTILE 128
#define KTILE 64
#define VSTR  72   // padded key-stride (bf16) of transposed V tile (144B rows)

union V16U { v16bf v; uint4 u[2]; };

__device__ __forceinline__ v16bf ld2x16B(const __bf16* p0, const __bf16* p1) {
  V16U r;
  r.u[0] = *(const uint4*)p0;
  r.u[1] = *(const uint4*)p1;
  return r.v;
}

__device__ __forceinline__ float fast_exp2(float x) {
#if __has_builtin(__builtin_amdgcn_exp2f)
  return __builtin_amdgcn_exp2f(x);   // bare v_exp_f32
#else
  return exp2f(x);
#endif
}
__device__ __forceinline__ float fast_rcp(float x) {
#if __has_builtin(__builtin_amdgcn_rcpf)
  return __builtin_amdgcn_rcpf(x);    // bare v_rcp_f32
#else
  return 1.0f / x;
#endif
}

__global__ __launch_bounds__(256)
void fa_causal_bf16_wmma_t(const float* __restrict__ Q,
                           const float* __restrict__ Kg,
                           const float* __restrict__ Vg,
                           float* __restrict__ Og)
{
  __shared__ __bf16 sK[2][KTILE * DK_];    // [key][dk] row-major, 2 x 8KB
  __shared__ __bf16 sVt[2][DK_ * VSTR];    // [dk][key] padded,   2 x 9KB

  const int tid  = threadIdx.x;
  const int w    = tid >> 5;
  const int lane = tid & 31;
  const int b    = blockIdx.y;
  const int qt0  = blockIdx.x * QTILE;
  const int qbase = qt0 + w * 16;          // this wave's 16 query rows
  const int col  = lane & 15;
  const int half = lane >> 4;
  const int hi8  = half * 8;
  const int hi16 = half * 16;

  // fold 1/sqrt(64) and log2(e) into Q -> exp2-domain softmax
  constexpr float QSCALE = 0.125f * 1.44269504088896340736f;

  // ---- Q as two B-fragments: lane=query col, elem t -> dk = f*32+hi16+t ----
  const float* Qp = Q + (size_t)(b * S_ + qbase + col) * DK_;
  v16bf qb0, qb1;
  #pragma unroll
  for (int t = 0; t < 16; ++t) {
    qb0[t] = (__bf16)(Qp[     hi16 + t] * QSCALE);
    qb1[t] = (__bf16)(Qp[32 + hi16 + t] * QSCALE);
  }

  // all-ones B fragment, made opaque so it is materialized once in VGPRs
  unsigned onebits = 0x3F803F80u;          // two packed bf16 1.0
  asm("" : "+s"(onebits));
  V16U ou;
  ou.u[0] = make_uint4(onebits, onebits, onebits, onebits);
  ou.u[1] = make_uint4(onebits, onebits, onebits, onebits);
  const v16bf ones = ou.v;

  v8f acc[4], accL;
  #pragma unroll
  for (int d = 0; d < 4; ++d) { v8f z = {}; acc[d] = z; }
  { v8f z = {}; accL = z; }
  float mcur = -INFINITY;                  // running max for this lane's query

  const int nblk = (qt0 + QTILE) / KTILE;  // 2*blockIdx.x + 2
  // K staging: thread -> one key row, 16 dk (two b128 LDS stores)
  const int skey = tid >> 2;
  const int sdk  = (tid & 3) * 16;
  // V staging: thread -> two adjacent key rows, 8 dk (eight b32 LDS stores)
  const int vkp  = (tid & 31) * 2;
  const int vdk  = (tid >> 5) * 8;
  const float* Kp = Kg + (size_t)(b * S_) * DK_;
  const float* Vp = Vg + (size_t)(b * S_) * DK_;

  // register-staged tile (filled for block blk+1 while computing blk)
  float4 kR0, kR1, kR2, kR3, vA0, vA1, vB0, vB1;

  auto load_tile = [&](int j0) {
    const float4* kg4 = (const float4*)(Kp + (size_t)(j0 + skey) * DK_ + sdk);
    kR0 = kg4[0]; kR1 = kg4[1]; kR2 = kg4[2]; kR3 = kg4[3];
    const float* vr0 = Vp + (size_t)(j0 + vkp) * DK_ + vdk;
    const float4* va  = (const float4*)vr0;
    const float4* vb4 = (const float4*)(vr0 + DK_);
    vA0 = va[0]; vA1 = va[1]; vB0 = vb4[0]; vB1 = vb4[1];
  };
  auto store_tile = [&](int buf) {
    union { __bf16 h[16]; uint4 u[2]; } pk;
    float kf[16] = {kR0.x,kR0.y,kR0.z,kR0.w, kR1.x,kR1.y,kR1.z,kR1.w,
                    kR2.x,kR2.y,kR2.z,kR2.w, kR3.x,kR3.y,kR3.z,kR3.w};
    #pragma unroll
    for (int i = 0; i < 16; ++i) pk.h[i] = (__bf16)kf[i];
    uint4* dstk = (uint4*)&sK[buf][skey * DK_ + sdk];
    dstk[0] = pk.u[0];
    dstk[1] = pk.u[1];

    float fa[8] = {vA0.x,vA0.y,vA0.z,vA0.w, vA1.x,vA1.y,vA1.z,vA1.w};
    float fb[8] = {vB0.x,vB0.y,vB0.z,vB0.w, vB1.x,vB1.y,vB1.z,vB1.w};
    #pragma unroll
    for (int i = 0; i < 8; ++i) {
      union { __bf16 h[2]; unsigned int u; } p2;
      p2.h[0] = (__bf16)fa[i];
      p2.h[1] = (__bf16)fb[i];
      *(unsigned int*)&sVt[buf][(vdk + i) * VSTR + vkp] = p2.u;
    }
  };

  load_tile(0);   // prologue: tile 0 into registers

  for (int blk = 0; blk < nblk; ++blk) {
    const int j0  = blk * KTILE;
    const int cur = blk & 1;

    store_tile(cur);       // safe vs compute(blk-2): see barrier argument
    __syncthreads();       // publish buffer `cur` (single barrier per block)

    if (blk + 1 < nblk) {  // stage next tile; latency hidden under compute
      load_tile(j0 + KTILE);
      if (blk + 2 < nblk) {
        __builtin_prefetch(Kp + (size_t)(j0 + 2*KTILE + skey) * DK_ + sdk, 0, 0);
        __builtin_prefetch(Vp + (size_t)(j0 + 2*KTILE + vkp)  * DK_ + vdk, 0, 0);
      }
    }

    // wave-uniform causal skip
    if (j0 <= qbase + 15) {
      // ---- S^T tiles: st[t] = K(16x64) x Q^T(64x16) ----
      v8f st[4];
      #pragma unroll
      for (int t = 0; t < 4; ++t) {
        const __bf16* kr = &sK[cur][(t * 16 + col) * DK_];
        v16bf a0 = ld2x16B(kr + hi16,      kr + hi16 + 8);   // dk  0..31
        v16bf a1 = ld2x16B(kr + 32 + hi16, kr + 40 + hi16);  // dk 32..63
        v8f c = {};
        c = __builtin_amdgcn_wmma_f32_16x16x32_bf16(false, a1, false, qb1,
                                                    (short)0, c, false, false);
        c = __builtin_amdgcn_wmma_f32_16x16x32_bf16(false, a0, false, qb0,
                                                    (short)0, c, false, false);
        st[t] = c;
      }

      // ---- causal mask: key j = j0 + t*16 + hi8 + i, query q = qbase+col ----
      if (j0 + KTILE - 1 > qbase) {
        const int basejq = j0 + hi8 - (qbase + col);
        #pragma unroll
        for (int t = 0; t < 4; ++t)
          #pragma unroll
          for (int i = 0; i < 8; ++i)
            if (basejq + t * 16 + i > 0) st[t][i] = -INFINITY;
      }

      // ---- in-lane max over 32 scores + one cross-half shuffle ----
      v8f t01, t23;
      #pragma unroll
      for (int i = 0; i < 8; ++i) {
        t01[i] = fmaxf(st[0][i], st[1][i]);
        t23[i] = fmaxf(st[2][i], st[3][i]);
      }
      #pragma unroll
      for (int i = 0; i < 8; ++i) t01[i] = fmaxf(t01[i], t23[i]);
      float mx = fmaxf(fmaxf(fmaxf(t01[0], t01[1]), fmaxf(t01[2], t01[3])),
                       fmaxf(fmaxf(t01[4], t01[5]), fmaxf(t01[6], t01[7])));
      mx = fmaxf(mx, __shfl_xor(mx, 16, 32));
      const float mnew = fmaxf(mcur, mx);

      // ---- rescale accumulators only if some lane's max moved ----
      if (__builtin_amdgcn_ballot_w32(mnew > mcur)) {
        const float alpha = fast_exp2(mcur - mnew);
        float ab[8];
        #pragma unroll
        for (int i = 0; i < 8; ++i) ab[i] = __shfl(alpha, hi8 + i, 32);
        #pragma unroll
        for (int i = 0; i < 8; ++i) {
          acc[0][i] *= ab[i]; acc[1][i] *= ab[i];
          acc[2][i] *= ab[i]; acc[3][i] *= ab[i];
          accL[i]   *= ab[i];
        }
      }
      mcur = mnew;

      // ---- p = exp2(s - m); P^T C-layout == P A-layout: register-only ----
      v16bf pa0, pa1;
      #pragma unroll
      for (int i = 0; i < 8; ++i) {
        pa0[i]     = (__bf16)fast_exp2(st[0][i] - mnew);
        pa0[i + 8] = (__bf16)fast_exp2(st[1][i] - mnew);
        pa1[i]     = (__bf16)fast_exp2(st[2][i] - mnew);
        pa1[i + 8] = (__bf16)fast_exp2(st[3][i] - mnew);
      }

      // ---- O += P x V (two 32-key passes); accL += P x ones (row sums) ----
      #pragma unroll
      for (int s = 0; s < 2; ++s) {
        const v16bf pa = s ? pa1 : pa0;
        #pragma unroll
        for (int d = 0; d < 4; ++d) {
          const __bf16* vr = &sVt[cur][(d * 16 + col) * VSTR + s * 32 + hi8];
          v16bf vb = ld2x16B(vr, vr + 16);
          acc[d] = __builtin_amdgcn_wmma_f32_16x16x32_bf16(false, pa, false, vb,
                                                           (short)0, acc[d],
                                                           false, false);
        }
        accL = __builtin_amdgcn_wmma_f32_16x16x32_bf16(false, pa, false, ones,
                                                       (short)0, accL,
                                                       false, false);
      }
    }
  }

  // ---- normalize (accL already in row layout) and store ----
  #pragma unroll
  for (int i = 0; i < 8; ++i) {
    const float inv = fast_rcp(accL[i]);
    const int q = qbase + hi8 + i;
    float* op = Og + (size_t)(b * S_ + q) * DK_;
    #pragma unroll
    for (int d = 0; d < 4; ++d)
      op[d * 16 + col] = acc[d][i] * inv;
  }
}

extern "C" void kernel_launch(void* const* d_in, const int* in_sizes, int n_in,
                              void* d_out, int out_size, void* d_ws, size_t ws_size,
                              hipStream_t stream) {
  const float* Q = (const float*)d_in[0];
  const float* K = (const float*)d_in[1];
  const float* V = (const float*)d_in[2];
  // d_in[3] causal mask: applied analytically in-kernel.
  float* O = (float*)d_out;
  (void)in_sizes; (void)n_in; (void)out_size; (void)d_ws; (void)ws_size;

  dim3 grid(S_ / QTILE, B_);   // 32 x 16 = 512 workgroups
  dim3 block(256);             // 8 wave32 waves
  fa_causal_bf16_wmma_t<<<grid, block, 0, stream>>>(Q, K, V, O);
}
